// MultiResolutionSpectrogram_24575802868105
// MI455X (gfx1250) — matrix operand order repaired and checked by reading, not compiled, not run on hardware
//
#include <hip/hip_runtime.h>
#include <hip/hip_bf16.h>
#include <math.h>

typedef __attribute__((ext_vector_type(16))) _Float16 v16h;
typedef __attribute__((ext_vector_type(8)))  _Float16 v8h;
typedef __attribute__((ext_vector_type(8)))  float    v8f;
typedef __attribute__((ext_vector_type(4)))  float    v4f;

#define T_SAMPLES 16384
#define NUM_BATCH 128
#define NUM_FREQS 257
#define NUM_TIMES 513

// Per-resolution constants (n_fft, F_s = n/2+1, N_s = T/hop+1, padded to mult-of-16)
static const int kNfft [4] = {  64, 128, 256, 512 };
static const int kFs   [4] = {  33,  65, 129, 257 };
static const int kNs   [4] = { 513, 257, 129,  65 };
static const int kNsPad[4] = { 528, 272, 144,  80 };
static const int kFpad [4] = {  48,  80, 144, 272 };

// ---------------------------------------------------------------------------
// 1) Build windowed frames (reflect pad, periodic Hann), f16, M-padded w/ zeros
// ---------------------------------------------------------------------------
__global__ void build_frames(const float* __restrict__ X,
                             _Float16* __restrict__ frames,
                             int n_fft, int Ns, int NsPad, long total)
{
    long e = (long)blockIdx.x * blockDim.x + threadIdx.x;
    if (e >= total) return;
    int  j   = (int)(e % n_fft);
    long rem = e / n_fft;
    int  t   = (int)(rem % NsPad);
    int  b   = (int)(rem / NsPad);

    _Float16 val = (_Float16)0.0f;
    if (t < Ns) {
        int hop = n_fft >> 1;              // pad == hop == n_fft/2
        int pos = t * hop + j - hop;
        if (pos < 0)          pos = -pos;                       // reflect left
        if (pos >= T_SAMPLES) pos = 2 * T_SAMPLES - 2 - pos;    // reflect right
        float w = 0.5f - 0.5f * __cosf(6.28318530717958647692f * (float)j / (float)n_fft);
        val = (_Float16)(X[(long)b * T_SAMPLES + pos] * w);
    }
    frames[e] = val;
}

// ---------------------------------------------------------------------------
// 2) Build transposed DFT basis: rows [0,Fpad) = cos, rows [Fpad,2*Fpad) = -sin
//    basisT[f2][k], k contiguous -> B fragments load as contiguous b128s.
// ---------------------------------------------------------------------------
__global__ void build_basis(_Float16* __restrict__ basisT,
                            int n_fft, int Fpad, long total)
{
    long e = (long)blockIdx.x * blockDim.x + threadIdx.x;
    if (e >= total) return;
    int k  = (int)(e % n_fft);
    int f2 = (int)(e / n_fft);
    int f  = (f2 < Fpad) ? f2 : (f2 - Fpad);
    int m  = (k * f) % n_fft;              // exact argument reduction
    float ph = 6.28318530717958647692f * (float)m / (float)n_fft;
    float s, c;
    __sincosf(ph, &s, &c);
    basisT[e] = (_Float16)((f2 < Fpad) ? c : -s);
}

// ---------------------------------------------------------------------------
// 3) WMMA GEMM. Block = 4 waves, all sharing one 16-frame M tile.
//    The 16 x NFFT A tile is contiguous in memory; stage it into LDS ONCE per
//    block with CDNA5 async global->LDS b128 copies (ASYNCcnt), then each wave
//    reads its A fragments from LDS (ds_load_b128) and B from global (b128).
//    Two accumulators (cos / sin) so power = (re^2+im^2)*invWin in-register.
//      A frag halves: K = k0 + (lane>>4)*8 + {0..7} and + 16
//      B frag halves: K = k0 + (lane>>4)*8 + {0..7} and + 16
//      C frag: lane holds N = lane&15, M = (lane>>4)*8 + i -> 8 consecutive t.
// ---------------------------------------------------------------------------
template <int NFFT>
__global__ __launch_bounds__(128)
void spec_gemm(const _Float16* __restrict__ frames,
               const _Float16* __restrict__ basisT,
               float* __restrict__ spec,
               int NsPad, int Fpad, int Ftiles, float invWin)
{
    __shared__ _Float16 ldsA[16 * NFFT];

    const int mtile = blockIdx.x;
    const int b     = blockIdx.z;

    // ---- cooperative async copy: 16*NFFT contiguous halves -> LDS ----
    const _Float16* Abase = frames + (long)b * ((long)NsPad * NFFT)
                          + (long)(mtile * 16) * NFFT;
    constexpr int CHUNKS = (16 * NFFT) / (128 * 8);   // b128 chunks per thread
#pragma unroll
    for (int c = 0; c < CHUNKS; ++c) {
        const int idx = threadIdx.x + c * 128;
        // generic pointer to __shared__: low 32 bits == LDS byte offset
        unsigned lds_off = (unsigned)(size_t)(&ldsA[idx * 8]);
        unsigned long long gaddr = (unsigned long long)(size_t)(Abase + idx * 8);
        asm volatile("global_load_async_to_lds_b128 %0, %1, off"
                     :: "v"(lds_off), "v"(gaddr) : "memory");
    }
    asm volatile("s_wait_asynccnt 0" ::: "memory");
    __syncthreads();

    const int wave  = threadIdx.x >> 5;
    const int lane  = threadIdx.x & 31;
    const int ftile = blockIdx.y * 4 + wave;          // wave-uniform
    if (ftile >= Ftiles) return;                      // uniform; after barrier

    const int hsel   = lane >> 4;   // 0 | 1
    const int laneLo = lane & 15;

    const _Float16* Arow = &ldsA[laneLo * NFFT + hsel * 8];
    const _Float16* Bc   = basisT + (long)(ftile * 16 + laneLo) * NFFT + hsel * 8;
    const _Float16* Bs   = Bc + (long)Fpad * NFFT;

    v8f cre = {};
    v8f cim = {};
#pragma unroll
    for (int k0 = 0; k0 < NFFT; k0 += 32) {
        v8h a0  = *(const v8h*)(Arow + k0);
        v8h a1  = *(const v8h*)(Arow + k0 + 16);
        v16h a  = __builtin_shufflevector(a0, a1, 0,1,2,3,4,5,6,7,8,9,10,11,12,13,14,15);
        v8h bc0 = *(const v8h*)(Bc + k0);
        v8h bc1 = *(const v8h*)(Bc + k0 + 16);
        v16h bc = __builtin_shufflevector(bc0, bc1, 0,1,2,3,4,5,6,7,8,9,10,11,12,13,14,15);
        v8h bs0 = *(const v8h*)(Bs + k0);
        v8h bs1 = *(const v8h*)(Bs + k0 + 16);
        v16h bs = __builtin_shufflevector(bs0, bs1, 0,1,2,3,4,5,6,7,8,9,10,11,12,13,14,15);
        cre = __builtin_amdgcn_wmma_f32_16x16x32_f16(false, a, false, bc, (short)0, cre, false, false);
        cim = __builtin_amdgcn_wmma_f32_16x16x32_f16(false, a, false, bs, (short)0, cim, false, false);
    }

    const int f     = ftile * 16 + laneLo;
    const int tBase = mtile * 16 + hsel * 8;
    float* outp = spec + (long)b * ((long)Fpad * NsPad) + (long)f * NsPad + tBase;

    v4f p0, p1;
#pragma unroll
    for (int i = 0; i < 4; ++i) p0[i] = (cre[i]     * cre[i]     + cim[i]     * cim[i])     * invWin;
#pragma unroll
    for (int i = 0; i < 4; ++i) p1[i] = (cre[i + 4] * cre[i + 4] + cim[i + 4] * cim[i + 4]) * invWin;
    *(v4f*)(outp)     = p0;
    *(v4f*)(outp + 4) = p1;
}

// ---------------------------------------------------------------------------
// 4) Nearest-index remap onto (257, 513) grid + running max across resolutions
// ---------------------------------------------------------------------------
__global__ void remap_max(const float* __restrict__ spec,
                          float* __restrict__ out,
                          int Fs, int Ns, int NsPad, long specBatch,
                          int first, long total)
{
    long e = (long)blockIdx.x * blockDim.x + threadIdx.x;
    if (e >= total) return;
    int  T   = (int)(e % NUM_TIMES);
    long rem = e / NUM_TIMES;
    int  F   = (int)(rem % NUM_FREQS);
    int  b   = (int)(rem / NUM_FREQS);

    int fi = (F * Fs) / NUM_FREQS;
    int ti = (T * Ns) / NUM_TIMES;
    float v = spec[(long)b * specBatch + (long)fi * NsPad + ti];
    out[e] = first ? v : fmaxf(out[e], v);
}

// ---------------------------------------------------------------------------
// Host driver: loop resolutions, reusing frames/spec workspace.
//   ws layout: [frames f16 <= 10.5MB][spec f32 <= 13.0MB][basis f16 <= 0.56MB]
// ---------------------------------------------------------------------------
extern "C" void kernel_launch(void* const* d_in, const int* in_sizes, int n_in,
                              void* d_out, int out_size, void* d_ws, size_t ws_size,
                              hipStream_t stream)
{
    (void)in_sizes; (void)n_in; (void)out_size; (void)ws_size;
    const float* X   = (const float*)d_in[0];
    float*       out = (float*)d_out;

    char* ws = (char*)d_ws;
    _Float16* frames = (_Float16*)(ws);
    float*    spec   = (float*)   (ws + 11l * 1024 * 1024);
    _Float16* basis  = (_Float16*)(ws + 25l * 1024 * 1024);

    for (int r = 0; r < 4; ++r) {
        const int n     = kNfft[r];
        const int Fs    = kFs[r];
        const int Ns    = kNs[r];
        const int NsPad = kNsPad[r];
        const int Fpad  = kFpad[r];

        const long fTot = (long)NUM_BATCH * NsPad * n;
        build_frames<<<(int)((fTot + 255) / 256), 256, 0, stream>>>(
            X, frames, n, Ns, NsPad, fTot);

        const long bTot = (long)2 * Fpad * n;
        build_basis<<<(int)((bTot + 255) / 256), 256, 0, stream>>>(
            basis, n, Fpad, bTot);

        const int Ftiles = Fpad / 16;
        dim3 g(NsPad / 16, (Ftiles + 3) / 4, NUM_BATCH);
        const float invWin = 8.0f / (3.0f * (float)n);
        switch (n) {
        case 64:  spec_gemm<64> <<<g, 128, 0, stream>>>(frames, basis, spec, NsPad, Fpad, Ftiles, invWin); break;
        case 128: spec_gemm<128><<<g, 128, 0, stream>>>(frames, basis, spec, NsPad, Fpad, Ftiles, invWin); break;
        case 256: spec_gemm<256><<<g, 128, 0, stream>>>(frames, basis, spec, NsPad, Fpad, Ftiles, invWin); break;
        default:  spec_gemm<512><<<g, 128, 0, stream>>>(frames, basis, spec, NsPad, Fpad, Ftiles, invWin); break;
        }

        const long oTot = (long)NUM_BATCH * NUM_FREQS * NUM_TIMES;
        remap_max<<<(int)((oTot + 255) / 256), 256, 0, stream>>>(
            spec, out, Fs, Ns, NsPad, (long)Fpad * NsPad, r == 0, oTot);
    }
}
